// GraphNet_9594956939811
// MI455X (gfx1250) — matrix operand order repaired and checked by reading, not compiled, not run on hardware
//
#include <hip/hip_runtime.h>
#include <hip/hip_bf16.h>

// ---------------------------------------------------------------------------
// GraphNet on gfx1250 (MI455X).  fp32 state in memory, fp16 WMMA matmuls
// (v_wmma_f32_16x16x32_f16) with fp32 accumulate.  Weight panels stored
// TRANSPOSED (64 x 256) f16: each lane's B fragment is picked from two 32-byte
// LDS loads (8 halves of each, table k order).  Panel staged into LDS with
// GLOBAL_LOAD_ASYNC_TO_LDS_B128 (ASYNCcnt) when available.  A fragments are
// gathered as address_space(1) raw-vector float4 -> global_load_b128.
// ---------------------------------------------------------------------------

typedef __attribute__((ext_vector_type(16))) _Float16 v16h;
typedef __attribute__((ext_vector_type(8)))  float    v8f;
typedef __attribute__((ext_vector_type(4)))  float    v4f;   // raw float4 (no HIP class)
typedef int v4i_b128 __attribute__((vector_size(16)));       // raw type for async-LDS builtin

#define Nn   8192
#define Mm   256
#define APMc 32
#define Ee   15616
#define Dd   64
#define FINc 16
#define EPMc 61     // edges per molecule (31 + 30)

#define GAS __attribute__((address_space(1)))
#define LAS __attribute__((address_space(3)))

#if __has_builtin(__builtin_amdgcn_global_load_async_to_lds_b128)
#define USE_ASYNC_LDS 1
#endif

// ---------------- small utility kernels ----------------

// convert W (K x 64 fp32, K=256 rows used) to TRANSPOSED f16 panel (64 x 256)
__global__ void k_f2h_t(const float* __restrict__ s, _Float16* __restrict__ d) {
    int i = blockIdx.x * 256 + threadIdx.x;     // over 256*64
    int k = i >> 6, c = i & 63;
    d[c * 256 + k] = (_Float16)s[i];
}

__global__ void k_zero(float* __restrict__ p, int n) {
    int i = blockIdx.x * 256 + threadIdx.x;
    if (i < n) p[i] = 0.0f;
}

// h_v = tanh(atoms @ W_fv + b_fv)   (8192x16)@(16x64), tiny: scalar fp32
__global__ void k_init_v(const float* __restrict__ atoms,
                         const float* __restrict__ W, const float* __restrict__ b,
                         float* __restrict__ hv, float* __restrict__ hv0) {
    int i = blockIdx.x * 256 + threadIdx.x;     // over N*D
    int n = i >> 6, d = i & 63;
    float acc = b[d];
#pragma unroll
    for (int k = 0; k < FINc; ++k) acc += atoms[n * FINc + k] * W[k * Dd + d];
    float t = tanhf(acc);
    hv[i] = t; hv0[i] = t;
}

// per-mol atom-feature sum (contiguous 32 atoms) then 16->64 matmul + tanh
__global__ void k_init_u(const float* __restrict__ atoms,
                         const float* __restrict__ W, const float* __restrict__ b,
                         float* __restrict__ hu, float* __restrict__ hu0) {
    __shared__ float s[FINc];
    int m = blockIdx.x, d = threadIdx.x;        // 256 blocks x 64 threads
    if (d < FINc) {
        float a = 0.f;
        for (int j = 0; j < APMc; ++j) a += atoms[(m * APMc + j) * FINc + d];
        s[d] = a;
    }
    __syncthreads();
    float acc = b[d];
#pragma unroll
    for (int k = 0; k < FINc; ++k) acc += s[k] * W[k * Dd + d];
    float t = tanhf(acc);
    hu[m * Dd + d] = t; hu0[m * Dd + d] = t;
}

// h_e = tanh(bond_order * W_fe + b_fe)  (rank-1)
__global__ void k_init_e(const float* __restrict__ adj,
                         const int* __restrict__ left, const int* __restrict__ right,
                         const float* __restrict__ We, const float* __restrict__ be,
                         float* __restrict__ he, float* __restrict__ he0) {
    int i = blockIdx.x * 256 + threadIdx.x;     // over E*D
    int e = i >> 6, d = i & 63;
    float bo = adj[(size_t)left[e] * Nn + right[e]];
    float t = tanhf(bo * We[d] + be[d]);
    he[i] = t; he0[i] = t;
}

// per-mol row-sum of h_u (u_e scalar term)
__global__ void k_usum(const float* __restrict__ hu, float* __restrict__ us) {
    int m = threadIdx.x;                        // 1 block x 256 threads
    float a = 0.f;
#pragma unroll
    for (int d = 0; d < Dd; ++d) a += hu[m * Dd + d];
    us[m] = a;
}

// scatter-add edge features to both incident atoms
__global__ void k_scatter(const float* __restrict__ he,
                          const int* __restrict__ left, const int* __restrict__ right,
                          float* __restrict__ dst) {
    int i = blockIdx.x * 256 + threadIdx.x;     // over E*D
    int e = i >> 6, d = i & 63;
    float v = he[i];
    atomicAdd(dst + (size_t)left[e]  * Dd + d, v);
    atomicAdd(dst + (size_t)right[e] * Dd + d, v);
}

// segment sum of h_e per molecule (61 contiguous edges)
__global__ void k_segE(const float* __restrict__ he, float* __restrict__ out) {
    int m = blockIdx.x, d = threadIdx.x;        // 256 x 64
    float a = 0.f;
    for (int j = 0; j < EPMc; ++j) a += he[((size_t)(m * EPMc + j) << 6) + d];
    out[m * Dd + d] = a;
}

// segment sum of h_v per molecule (32 contiguous atoms)
__global__ void k_segV(const float* __restrict__ hv, float* __restrict__ out) {
    int m = blockIdx.x, d = threadIdx.x;        // 256 x 64
    float a = 0.f;
#pragma unroll
    for (int j = 0; j < APMc; ++j) a += hv[((size_t)(m * APMc + j) << 6) + d];
    out[m * Dd + d] = a;
}

// ---------------- unified WMMA phi kernel ----------------
// out[tile rows, 64] = tanh( [A0|A1|A2|A3] (K=256) @ W(256x64)  + bias
//                             (+ u_e * W_extra  for mode 0) )
// mode 0 (phi_e): A2/A3 rows gathered via left/right edge indices, u_e term on
// mode 1 (phi_v): A3 row = atom_row >> 5 (h_u[mol])
// mode 2 (phi_u): all sources row-aligned
__global__ __launch_bounds__(128) void k_phi(
    int mode,
    const float* __restrict__ s0, const float* __restrict__ s1,
    const float* __restrict__ s2, const float* __restrict__ s3,
    const int* __restrict__ left, const int* __restrict__ right,
    const _Float16* __restrict__ Wg,        // 64 x 256 f16 (TRANSPOSED panel)
    const float* __restrict__ Wextra,       // mode 0: W_pe row 256 (fp32)
    const float* __restrict__ bias,
    const float* __restrict__ uSum,
    float* __restrict__ out)
{
    __shared__ _Float16 Wl[Dd * 256];       // 32 KB transposed weight panel
    __shared__ float sX[Dd];
    __shared__ float sB[Dd];

    const int tid = threadIdx.x;

    // ---- stage the f16 weight panel into LDS ----
#ifdef USE_ASYNC_LDS
    {
        // gfx1250 async copy: GLOBAL_LOAD_ASYNC_TO_LDS_B128, tracked by ASYNCcnt
        GAS v4i_b128* g = (GAS v4i_b128*)Wg;
        LAS v4i_b128* l = (LAS v4i_b128*)Wl;
#pragma unroll
        for (int i = 0; i < 16; ++i)
            __builtin_amdgcn_global_load_async_to_lds_b128(
                g + tid + i * 128, l + tid + i * 128, 0, 0);
        asm volatile("s_wait_asynccnt 0x0" ::: "memory");
    }
#else
    {
        const uint4* g = (const uint4*)Wg;
        uint4* l = (uint4*)Wl;
#pragma unroll
        for (int i = 0; i < 16; ++i) l[tid + i * 128] = g[tid + i * 128];
    }
#endif
    if (tid < Dd) { sX[tid] = Wextra[tid]; sB[tid] = bias[tid]; }
    __syncthreads();

    const int wave = tid >> 5;
    const int lane = tid & 31;
    const int tile = blockIdx.x * 4 + wave;     // 16-row tile per wave
    const int mrow = (tile << 4) + (lane & 15); // this lane's A row
    const int hi   = lane >> 4;
    const int cn   = lane & 15;

    GAS const float* p[4];
    p[0] = (GAS const float*)(s0 + (size_t)mrow * Dd);
    p[1] = (GAS const float*)(s1 + (size_t)mrow * Dd);
    if (mode == 0) {
        p[2] = (GAS const float*)(s2 + (size_t)left[mrow]  * Dd);
        p[3] = (GAS const float*)(s3 + (size_t)right[mrow] * Dd);
    } else if (mode == 1) {
        p[2] = (GAS const float*)(s2 + (size_t)mrow * Dd);
        p[3] = (GAS const float*)(s3 + (size_t)(mrow >> 5) * Dd);  // h_u[atom_mol]
    } else {
        p[2] = (GAS const float*)(s2 + (size_t)mrow * Dd);
        p[3] = (GAS const float*)(s3 + (size_t)mrow * Dd);
    }

    v8f acc[4] = {};                            // 4 N-tiles of 16

#pragma unroll
    for (int kc = 0; kc < 8; ++kc) {            // K = 8 x 32
        GAS const float* ps = p[kc >> 1];       // concat source for this chunk
        const int koff = ((kc & 1) << 5) + (hi << 3);
        // A fragment: two contiguous 32B runs -> 4x global_load_b128
        GAS const v4f* q = (GAS const v4f*)(ps + koff);
        v4f f0 = q[0];
        v4f f1 = q[1];
        v4f f2 = q[4];                          // koff + 16
        v4f f3 = q[5];
        v16h a;
        a[0]  = (_Float16)f0.x; a[1]  = (_Float16)f0.y;
        a[2]  = (_Float16)f0.z; a[3]  = (_Float16)f0.w;
        a[4]  = (_Float16)f1.x; a[5]  = (_Float16)f1.y;
        a[6]  = (_Float16)f1.z; a[7]  = (_Float16)f1.w;
        a[8]  = (_Float16)f2.x; a[9]  = (_Float16)f2.y;
        a[10] = (_Float16)f2.z; a[11] = (_Float16)f2.w;
        a[12] = (_Float16)f3.x; a[13] = (_Float16)f3.y;
        a[14] = (_Float16)f3.z; a[15] = (_Float16)f3.w;

        const int kb = (kc << 5);               // chunk K base; lane takes k 8h..8h+7, 16+8h..16+8h+7
#pragma unroll
        for (int t = 0; t < 4; ++t) {
            // B fragment: two 32B aligned loads of the transposed panel, table-order select
            v16h b0 = *(LAS const v16h*)(LAS const _Float16*)(Wl + ((t << 4) + cn) * 256 + kb);
            v16h b1 = *(LAS const v16h*)(LAS const _Float16*)(Wl + ((t << 4) + cn) * 256 + kb + 16);
            v16h b = hi ? __builtin_shufflevector(b0, b1, 8, 9, 10, 11, 12, 13, 14, 15, 24, 25, 26, 27, 28, 29, 30, 31)
                        : __builtin_shufflevector(b0, b1, 0, 1, 2, 3, 4, 5, 6, 7, 16, 17, 18, 19, 20, 21, 22, 23);
            acc[t] = __builtin_amdgcn_wmma_f32_16x16x32_f16(
                false, a, false, b, (short)0, acc[t], false, false);
        }
    }

    // epilogue: bias (+ rank-1 u_e column for phi_e), tanh, store fp32
    GAS float* og = (GAS float*)out;
#pragma unroll
    for (int t = 0; t < 4; ++t) {
#pragma unroll
        for (int r = 0; r < 8; ++r) {
            int orow = (tile << 4) + (hi << 3) + r;     // C layout M
            float v = acc[t][r] + sB[(t << 4) + cn];
            if (mode == 0)
                v += uSum[left[orow] >> 5] * sX[(t << 4) + cn];
            og[(size_t)orow * Dd + (t << 4) + cn] = tanhf(v);
        }
    }
}

// ---------------- launcher ----------------

extern "C" void kernel_launch(void* const* d_in, const int* in_sizes, int n_in,
                              void* d_out, int out_size, void* d_ws, size_t ws_size,
                              hipStream_t stream)
{
    const float* atoms = (const float*)d_in[0];
    const float* adj   = (const float*)d_in[1];
    const int*   left  = (const int*)d_in[2];
    const int*   right = (const int*)d_in[3];
    // d_in[4], d_in[5]: atom_in_mol / bond_in_mol masks — unneeded (contiguous layout)
    const float* W_fe = (const float*)d_in[6];
    const float* b_fe = (const float*)d_in[7];
    const float* W_fv = (const float*)d_in[8];
    const float* b_fv = (const float*)d_in[9];
    const float* W_fu = (const float*)d_in[10];
    const float* b_fu = (const float*)d_in[11];
    const float* W_pe = (const float*)d_in[12];   // (257,64)
    const float* b_pe = (const float*)d_in[13];
    const float* W_pv = (const float*)d_in[14];   // (256,64)
    const float* b_pv = (const float*)d_in[15];
    const float* W_pu = (const float*)d_in[16];   // (256,64)
    const float* b_pu = (const float*)d_in[17];

    char* ws = (char*)d_ws;
    size_t off = 0;
    auto take = [&](size_t bytes) -> char* {
        char* r = ws + off;
        off = (off + bytes + 255) & ~(size_t)255;
        return r;
    };

    float*    hE     = (float*)take((size_t)Ee * Dd * 4);
    float*    hE0    = (float*)take((size_t)Ee * Dd * 4);
    float*    hV     = (float*)take((size_t)Nn * Dd * 4);
    float*    hV0    = (float*)take((size_t)Nn * Dd * 4);
    float*    hU     = (float*)take((size_t)Mm * Dd * 4);
    float*    hU0    = (float*)take((size_t)Mm * Dd * 4);
    float*    hEbarI = (float*)take((size_t)Nn * Dd * 4);
    float*    hEbar  = (float*)take((size_t)Mm * Dd * 4);
    float*    hVbar  = (float*)take((size_t)Mm * Dd * 4);
    float*    uSum   = (float*)take((size_t)Mm * 4);
    _Float16* WpeH   = (_Float16*)take((size_t)Dd * 256 * 2); // transposed panels
    _Float16* WpvH   = (_Float16*)take((size_t)Dd * 256 * 2);
    _Float16* WpuH   = (_Float16*)take((size_t)Dd * 256 * 2);

    // ---- weight conversion (f32 -> transposed f16 64x256 panels) ----
    k_f2h_t<<<256 * Dd / 256, 256, 0, stream>>>(W_pe, WpeH);
    k_f2h_t<<<256 * Dd / 256, 256, 0, stream>>>(W_pv, WpvH);
    k_f2h_t<<<256 * Dd / 256, 256, 0, stream>>>(W_pu, WpuH);

    // ---- initial embeddings ----
    k_init_v<<<Nn * Dd / 256, 256, 0, stream>>>(atoms, W_fv, b_fv, hV, hV0);
    k_init_u<<<Mm, Dd, 0, stream>>>(atoms, W_fu, b_fu, hU, hU0);
    k_init_e<<<Ee * Dd / 256, 256, 0, stream>>>(adj, left, right, W_fe, b_fe, hE, hE0);
    k_usum<<<1, Mm, 0, stream>>>(hU, uSum);

    const float* WpeRow256 = W_pe + 256 * Dd;   // the u_e column of W_pe

    for (int it = 0; it < 3; ++it) {
        // phi_e : (15616 x 256) @ (256 x 64) + u_e rank-1 + bias, tanh
        k_phi<<<Ee / 16 / 4, 128, 0, stream>>>(
            0, hE, hE0, hV, hV, left, right, WpeH, WpeRow256, b_pe, uSum, hE);

        // rho_e_v : scatter-add edges to atoms
        k_zero<<<Nn * Dd / 256, 256, 0, stream>>>(hEbarI, Nn * Dd);
        k_scatter<<<Ee * Dd / 256, 256, 0, stream>>>(hE, left, right, hEbarI);

        // rho_e_u : per-mol edge sum (uses updated h_e)
        k_segE<<<Mm, Dd, 0, stream>>>(hE, hEbar);

        // phi_v : (8192 x 256) @ (256 x 64) + bias, tanh  (h_u still old)
        k_phi<<<Nn / 16 / 4, 128, 0, stream>>>(
            1, hV, hV0, hEbarI, hU, left, right, WpvH, b_pv, b_pv, uSum, hV);

        // rho_v_u : per-mol atom sum (uses updated h_v)
        k_segV<<<Mm, Dd, 0, stream>>>(hV, hVbar);

        // phi_u : (256 x 256) @ (256 x 64) + bias, tanh
        k_phi<<<Mm / 16 / 4, 128, 0, stream>>>(
            2, hU, hU0, hEbar, hVbar, left, right, WpuH, b_pu, b_pu, uSum, hU);

        // refresh u_e scalars for next round's phi_e
        k_usum<<<1, Mm, 0, stream>>>(hU, uSum);
    }

    // final edge representations -> output
    (void)hipMemcpyAsync(d_out, hE, (size_t)Ee * Dd * sizeof(float),
                         hipMemcpyDeviceToDevice, stream);
    (void)in_sizes; (void)n_in; (void)out_size; (void)ws_size;
}
